// UnifiedRecEncoder_4776003633235
// MI455X (gfx1250) — compile-verified, matrix operand order checked by reading, not run on hardware
//
#include <hip/hip_runtime.h>
#include <hip/hip_bf16.h>
#include <math.h>

typedef __attribute__((ext_vector_type(16))) __bf16 v16bf;
typedef __attribute__((ext_vector_type(8)))  __bf16 v8bf;
typedef __attribute__((ext_vector_type(8)))  float  v8f;
typedef __attribute__((ext_vector_type(4)))  unsigned int v4u;
typedef __attribute__((ext_vector_type(8)))  int v8i;
typedef __attribute__((ext_vector_type(4)))  int v4i;

#define DMODEL 512

__device__ __forceinline__ v8f wmma_bf16(v16bf a, v16bf b, v8f c) {
  return __builtin_amdgcn_wmma_f32_16x16x32_bf16(false, a, false, b, (short)0, c,
                                                 false, false);
}

__device__ __forceinline__ v8f zero8() {
  v8f z;
#pragma unroll
  for (int e = 0; e < 8; ++e) z[e] = 0.f;
  return z;
}

// A-fragment (16 rows x 32 K) from a row-major bf16 tile (contiguous in K).
// Per ISA 7.12.2: lane = row (mod 16); lanes 0-15 hold K 0-7,16-23,
// lanes 16-31 hold K 8-15,24-31. Also serves as the B-fragment of an
// operand stored transposed (row = N, contiguous in K).
__device__ __forceinline__ v16bf frag_rowmajor(const __bf16* base, int ld) {
  const int lane = threadIdx.x & 31;
  const int m = lane & 15;
  const int kh = (lane >> 4) << 3;
  const __bf16* r = base + m * ld + kh;
  v16bf f;
#pragma unroll
  for (int e = 0; e < 8; ++e) f[e] = r[e];
#pragma unroll
  for (int e = 0; e < 8; ++e) f[8 + e] = r[16 + e];
  return f;
}

// B-fragment (32 K x 16 cols) from a row-major (K x N) bf16 tile (strided).
__device__ __forceinline__ v16bf frag_b(const __bf16* base, int ld) {
  const int lane = threadIdx.x & 31;
  const int n = lane & 15;
  const int kh = (lane >> 4) << 3;
  v16bf f;
#pragma unroll
  for (int e = 0; e < 8; ++e) f[e] = base[(kh + e) * ld + n];
#pragma unroll
  for (int e = 0; e < 8; ++e) f[8 + e] = base[(kh + 16 + e) * ld + n];
  return f;
}

// ---------------------------------------------------------------------------
// Tensor Data Mover: 2D bf16 tile load, global -> LDS, with LDS row padding.
// D# group0: count=1, lds_addr, global_addr, type=2.
// D# group1: data_size=1(2B), pad_enable, pad_interval/amount, dims/strides.
// Issued by one wave; completion via s_wait_tensorcnt + workgroup barrier.
// Static LDS block is assumed to start at group-segment offset 0.
// ---------------------------------------------------------------------------
__device__ __forceinline__ void tdm_load_2d_bf16(
    unsigned lds_byte_off, const void* gaddr, unsigned tile_d0,
    unsigned tile_d1, unsigned tensor_d0, unsigned tensor_d1, unsigned stride0,
    unsigned pad_interval, unsigned pad_amount) {
  const unsigned long long ga = (unsigned long long)gaddr;
  v4u g0;
  g0[0] = 1u;                                // count=1, user descriptor
  g0[1] = lds_byte_off;                      // lds_addr
  g0[2] = (unsigned)ga;                      // global_addr[31:0]
  g0[3] = (unsigned)(ga >> 32) | (2u << 30); // global_addr[56:32] | type=2
  v8i g1;
  g1[0] = (int)((1u << 16) |                 // data_size = 2 bytes
                (1u << 20) |                 // pad_enable
                (pad_interval << 22) | (pad_amount << 25));
  g1[1] = (int)((tensor_d0 & 0xFFFFu) << 16);              // dim0[15:0]@63:48
  g1[2] = (int)((tensor_d0 >> 16) | ((tensor_d1 & 0xFFFFu) << 16));
  g1[3] = (int)((tensor_d1 >> 16) | (tile_d0 << 16));
  g1[4] = (int)(tile_d1 & 0xFFFFu);          // tile_dim1 (tile_dim2 = 0)
  g1[5] = (int)stride0;                      // tensor_dim0_stride[31:0]
  g1[6] = 0;
  g1[7] = 0;
  const v4i z4 = {0, 0, 0, 0};
#if defined(__clang_major__) && __clang_major__ >= 23
  const v8i z8 = {0, 0, 0, 0, 0, 0, 0, 0};
  __builtin_amdgcn_tensor_load_to_lds(g0, g1, z4, z4, z8, 0);
#else
  __builtin_amdgcn_tensor_load_to_lds(g0, g1, z4, z4, 0);
#endif
}

// ---------------------------------------------------------------------------
// Tiled GEMM: C(MxN) = act(A(MxK,f32) * W(KxN) + bias) [+ res]
// W supplied PRE-TRANSPOSED as WT (N x K, bf16) so both LDS operands are
// contiguous in K -> all fragment reads are ds_load_b128 pairs.
// B tiles staged by TDM (wave 0), A tiles cooperatively converted f32->bf16.
// Block tile 128x128, K-step 32, 8 waves x (32x64). M%128==N%128==K%32==0.
// ---------------------------------------------------------------------------
template <bool RELU, bool RES, bool BF16OUT>
__global__ __launch_bounds__(256) void gemm_kernel(
    const float* __restrict__ A, const __bf16* __restrict__ WT,
    const float* __restrict__ bias, const float* __restrict__ res,
    void* __restrict__ Cv, int M, int N, int K) {
  __shared__ __align__(16) char gsm[40960];
  __bf16* As = (__bf16*)gsm;             // 2 x (128 rows x 40) A tiles
  __bf16* Bt = (__bf16*)(gsm + 20480);   // 2 x (128 rows x 40) B^T tiles
  const unsigned BT_OFF = 20480;

  const int tid = threadIdx.x;
  const int m0 = blockIdx.y << 7;
  const int n0 = blockIdx.x << 7;
  const int wid = tid >> 5;
  const int wm = (wid & 3) << 5;
  const int wn = (wid >> 2) << 6;

  v8f acc[2][4];
#pragma unroll
  for (int i = 0; i < 2; ++i)
#pragma unroll
    for (int j = 0; j < 4; ++j) acc[i][j] = zero8();

  auto loadA = [&](int buf, int k0) {
#pragma unroll
    for (int i = 0; i < 4; ++i) {
      const int lin = tid + (i << 8);
      const int row = lin >> 3;
      const int c4 = (lin & 7) << 2;
      const float4 v4 = *reinterpret_cast<const float4*>(
          A + (size_t)(m0 + row) * K + k0 + c4);
      __bf16* d = As + buf * 5120 + row * 40 + c4;
      d[0] = (__bf16)v4.x;
      d[1] = (__bf16)v4.y;
      d[2] = (__bf16)v4.z;
      d[3] = (__bf16)v4.w;
    }
  };
  auto issueB = [&](int buf, int k0) {
    // 128 rows x 32 bf16 (64B rows), LDS padded +16B -> ld 40
    tdm_load_2d_bf16(BT_OFF + (unsigned)buf * 10240, WT + (size_t)n0 * K + k0,
                     32, 128, 32, 128, (unsigned)K, 3, 3);
  };

  const int nk = K >> 5;
  if (wid == 0) issueB(0, 0);
  loadA(0, 0);
  if (wid == 0) __builtin_amdgcn_s_wait_tensorcnt(0);
  __syncthreads();
  for (int kt = 0; kt < nk; ++kt) {
    const int cur = kt & 1;
    if (kt + 1 < nk) {
      if (wid == 0) issueB(cur ^ 1, (kt + 1) << 5);
      loadA(cur ^ 1, (kt + 1) << 5);
    }
    if (kt + 2 < nk)  // gfx1250 global_prefetch_b8 on upcoming weight tile
      __builtin_prefetch(WT + (size_t)n0 * K + ((kt + 2) << 5), 0, 1);
    v16bf af[2];
#pragma unroll
    for (int i = 0; i < 2; ++i)
      af[i] = frag_rowmajor(As + cur * 5120 + (wm + (i << 4)) * 40, 40);
    v16bf bfr[4];
#pragma unroll
    for (int j = 0; j < 4; ++j)
      bfr[j] = frag_rowmajor(Bt + cur * 5120 + (wn + (j << 4)) * 40, 40);
#pragma unroll
    for (int i = 0; i < 2; ++i)
#pragma unroll
      for (int j = 0; j < 4; ++j)
        acc[i][j] = wmma_bf16(af[i], bfr[j], acc[i][j]);
    if (wid == 0 && kt + 1 < nk) __builtin_amdgcn_s_wait_tensorcnt(0);
    __syncthreads();
  }

  const int lane = tid & 31;
  const int col0 = lane & 15;
  const int rh = (lane >> 4) << 3;
  float* Cf = (float*)Cv;
  __bf16* Cb = (__bf16*)Cv;
#pragma unroll
  for (int i = 0; i < 2; ++i)
#pragma unroll
    for (int j = 0; j < 4; ++j)
#pragma unroll
      for (int r = 0; r < 8; ++r) {
        const int row = m0 + wm + (i << 4) + rh + r;
        const int col = n0 + wn + (j << 4) + col0;
        float v = acc[i][j][r];
        if (bias) v += bias[col];
        if (RELU) v = fmaxf(v, 0.f);
        if (RES) v += res[(size_t)row * N + col];
        if (BF16OUT)
          Cb[(size_t)row * N + col] = (__bf16)v;
        else
          Cf[(size_t)row * N + col] = v;
      }
}

// ---------------------------------------------------------------------------
// Fused causal attention, one batch element per block, Q/K/V in bf16.
// Scores via WMMA with K chunks TDM-staged to LDS (tensor_dim1=L gives free
// zero-fill), wave softmax via __shfl_xor, normalized P -> per-wave LDS,
// O = P V via WMMA with V chunks TDM-staged. O written f32.
// Query rows >= L use a CLAMPED row address (branchless vector loads); their
// scores are garbage but finite, and the softmax writes zero P for them.
// ---------------------------------------------------------------------------
__global__ __launch_bounds__(256) void attention_kernel(
    const __bf16* __restrict__ Qb, const __bf16* __restrict__ Kb,
    const __bf16* __restrict__ Vb, float* __restrict__ O, int L, float scale) {
  __shared__ __align__(16) char smem[37632];
  __bf16* Ks = (__bf16*)smem;            // 80 x 40  (TDM dest, off 0)
  __bf16* Vs = (__bf16*)(smem + 6400);   // 32 x 72  (TDM dest, off 6400)
  __bf16* Ps = (__bf16*)(smem + 11008);  // 8 x 16 x 104

  const int tid = threadIdx.x;
  const int w = tid >> 5;
  const int lane = tid & 31;
  const int col0 = lane & 15;
  const int rh = (lane >> 4) << 3;
  const size_t base = (size_t)blockIdx.x * (size_t)L * DMODEL;
  const int qbase = w << 4;
  const bool active = qbase < L;

  // zero P scratch with dword stores (26624 bytes)
  for (int i = tid; i < 6656; i += 256) ((int*)Ps)[i] = 0;

  // ---- scores S = Q K^T ----
  v8f sc[5];
#pragma unroll
  for (int f = 0; f < 5; ++f) sc[f] = zero8();

  // branchless Q row: clamp to last valid row (finite data; masked later)
  const int qrow_raw = qbase + (lane & 15);
  const int qrow = (qrow_raw < L) ? qrow_raw : (L - 1);
  const __bf16* qrow_ptr = Qb + base + (size_t)qrow * DMODEL + rh;

  for (int dc = 0; dc < DMODEL / 32; ++dc) {
    __syncthreads();
    if (w == 0) {
      // 80 rows x 32 bf16 from (L x 512); rows >= L zero-filled by TDM
      tdm_load_2d_bf16(0, Kb + base + dc * 32, 32, 80, 32, (unsigned)L,
                       DMODEL, 3, 3);
      __builtin_amdgcn_s_wait_tensorcnt(0);
    }
    __syncthreads();
    if (active) {
      // two 16B vector loads -> global_load_b128 pair
      const v8bf lo = *(const v8bf*)(qrow_ptr + dc * 32);
      const v8bf hi = *(const v8bf*)(qrow_ptr + dc * 32 + 16);
      v16bf qa;
#pragma unroll
      for (int e = 0; e < 8; ++e) qa[e] = lo[e];
#pragma unroll
      for (int e = 0; e < 8; ++e) qa[8 + e] = hi[e];
#pragma unroll
      for (int f = 0; f < 5; ++f) {
        v16bf kb = frag_rowmajor(Ks + (f << 4) * 40, 40);  // K^T B-fragment
        sc[f] = wmma_bf16(qa, kb, sc[f]);
      }
    }
  }
  __syncthreads();

  // ---- causal softmax ----
  if (active) {
#pragma unroll
    for (int r = 0; r < 8; ++r) {
      const int row = qbase + rh + r;
      float mx = -3.0e38f;
#pragma unroll
      for (int f = 0; f < 5; ++f) {
        const int col = (f << 4) + col0;
        float s = sc[f][r] * scale;
        if (col > row || col >= L) s = -3.0e38f;
        sc[f][r] = s;
        mx = fmaxf(mx, s);
      }
#pragma unroll
      for (int o = 1; o < 16; o <<= 1) mx = fmaxf(mx, __shfl_xor(mx, o, 32));
      float sum = 0.f;
#pragma unroll
      for (int f = 0; f < 5; ++f) {
        const float p = __expf(sc[f][r] - mx);
        sc[f][r] = p;
        sum += p;
      }
#pragma unroll
      for (int o = 1; o < 16; o <<= 1) sum += __shfl_xor(sum, o, 32);
      const float inv = (row < L) ? (1.0f / sum) : 0.f;
#pragma unroll
      for (int f = 0; f < 5; ++f)
        Ps[((w << 4) + rh + r) * 104 + (f << 4) + col0] =
            (__bf16)(sc[f][r] * inv);
    }
  }

  // ---- O = P V ----
  for (int dout = 0; dout < DMODEL / 64; ++dout) {
    v8f oa[4];
#pragma unroll
    for (int j = 0; j < 4; ++j) oa[j] = zero8();
    for (int ks = 0; ks < 3; ++ks) {
      __syncthreads();
      if (w == 0) {
        const unsigned td1 =
            (L > (ks << 5)) ? (unsigned)(L - (ks << 5)) : 1u;
        // 32 rows x 64 bf16 (128B rows) -> ld 72 (pad +16B); rows beyond
        // valid range zero-filled (P is zero there anyway).
        tdm_load_2d_bf16(6400,
                         Vb + base + (size_t)(ks << 5) * DMODEL + (dout << 6),
                         64, 32, 64, td1, DMODEL, 4, 3);
        __builtin_amdgcn_s_wait_tensorcnt(0);
      }
      __syncthreads();
      if (active) {
        v16bf pa = frag_rowmajor(Ps + (w << 4) * 104 + (ks << 5), 104);
#pragma unroll
        for (int j = 0; j < 4; ++j) {
          v16bf vbf = frag_b(Vs + (j << 4), 72);
          oa[j] = wmma_bf16(pa, vbf, oa[j]);
        }
      }
    }
    if (active) {
#pragma unroll
      for (int j = 0; j < 4; ++j)
#pragma unroll
        for (int r = 0; r < 8; ++r) {
          const int row = qbase + rh + r;
          if (row < L) {
            const int col = (dout << 6) + (j << 4) + col0;
            O[base + (size_t)row * DMODEL + col] = oa[j][r];
          }
        }
    }
  }
}

// ---------------------------------------------------------------------------
// Glue kernels
// ---------------------------------------------------------------------------
// d (N x K) = transpose of s (K x N), f32 -> bf16
__global__ void cvtT_bf16_kernel(const float* __restrict__ s,
                                 __bf16* __restrict__ d, int K, int N) {
  const int i = blockIdx.x * 256 + threadIdx.x;
  if (i < K * N) {
    const int k = i % K;
    const int nn = i / K;
    d[i] = (__bf16)s[(size_t)k * N + nn];
  }
}

__global__ void build_combined_kernel(const float* __restrict__ hist,
                                      const float* __restrict__ act,
                                      float* __restrict__ outb, int n) {
  const int i = blockIdx.x * 256 + threadIdx.x;
  if (i >= n) return;
  const int d = i & 511;
  const int rem = i >> 9;
  const int r = rem % 65;
  const int b = rem / 65;
  outb[i] = (r < 64) ? hist[((size_t)b * 64 + r) * 512 + d]
                     : act[(size_t)b * 512 + d];
}

__global__ void build_seq_kernel(const float* __restrict__ hist,
                                 const float* __restrict__ tokemb,
                                 const int* __restrict__ toks,
                                 float* __restrict__ outb, int n) {
  const int i = blockIdx.x * 256 + threadIdx.x;
  if (i >= n) return;
  const int d = i & 511;
  const int rem = i >> 9;
  const int r = rem % 72;
  const int b = rem / 72;
  if (r < 64)
    outb[i] = hist[((size_t)b * 64 + r) * 512 + d];
  else
    outb[i] = tokemb[(size_t)toks[b * 8 + (r - 64)] * 512 + d];
}

__global__ void extract_rows_kernel(const float* __restrict__ src,
                                    float* __restrict__ dst, int Lsrc, int r0,
                                    int nrows, int n) {
  const int i = blockIdx.x * 256 + threadIdx.x;
  if (i >= n) return;
  const int d = i & 511;
  const int rem = i >> 9;
  const int t = rem % nrows;
  const int b = rem / nrows;
  dst[i] = src[((size_t)b * Lsrc + r0 + t) * 512 + d];
}

__global__ void concat2_kernel(const float* __restrict__ a,
                               const float* __restrict__ b2,
                               float* __restrict__ outb, int da, int db,
                               int n) {
  const int i = blockIdx.x * 256 + threadIdx.x;
  if (i >= n) return;
  const int c = i % (da + db);
  const int b = i / (da + db);
  outb[i] = (c < da) ? a[(size_t)b * da + c] : b2[(size_t)b * db + (c - da)];
}

// Gathered item rows ARE the transposed (N x K) operand: egt[n][k]
__global__ void gather_items_kernel(const float* __restrict__ embs,
                                    const int* __restrict__ ids,
                                    __bf16* __restrict__ outb) {
  const int i = blockIdx.x * 256 + threadIdx.x;
  if (i < 512 * 512) {
    const int nn = i >> 9;
    const int k = i & 511;
    outb[i] = (__bf16)embs[(size_t)ids[nn] * 512 + k];
  }
}

__global__ void zero_out_kernel(float* o) { o[0] = 0.f; }

__global__ __launch_bounds__(256) void nll_rows_kernel(
    const float* __restrict__ logits, const int* __restrict__ labels, int Vdim,
    float wgt, float* __restrict__ outb) {
  const int row = blockIdx.x;
  const float* lrow = logits + (size_t)row * Vdim;
  __shared__ float red[256];
  float mx = -3.4e38f;
  for (int j = threadIdx.x; j < Vdim; j += 256) mx = fmaxf(mx, lrow[j]);
  red[threadIdx.x] = mx;
  __syncthreads();
  for (int s = 128; s > 0; s >>= 1) {
    if (threadIdx.x < s)
      red[threadIdx.x] = fmaxf(red[threadIdx.x], red[threadIdx.x + s]);
    __syncthreads();
  }
  mx = red[0];
  __syncthreads();
  float sum = 0.f;
  for (int j = threadIdx.x; j < Vdim; j += 256) sum += __expf(lrow[j] - mx);
  red[threadIdx.x] = sum;
  __syncthreads();
  for (int s = 128; s > 0; s >>= 1) {
    if (threadIdx.x < s) red[threadIdx.x] += red[threadIdx.x + s];
    __syncthreads();
  }
  if (threadIdx.x == 0) {
    const int lab = labels ? labels[row] : row;
    atomicAdd(outb, (logf(red[0]) + mx - lrow[lab]) * wgt);
  }
}

// ---------------------------------------------------------------------------
// Host orchestration
// ---------------------------------------------------------------------------
extern "C" void kernel_launch(void* const* d_in, const int* in_sizes, int n_in,
                              void* d_out, int out_size, void* d_ws,
                              size_t ws_size, hipStream_t stream) {
  (void)in_sizes; (void)n_in; (void)out_size; (void)ws_size;
  constexpr int Bn = 512, S = 64, T = 8, D = 512, U = 256, V = 1024, FF = 2048;

  const float* hist_in = (const float*)d_in[0];
  const float* userf   = (const float*)d_in[1];
  const float* actenc  = (const float*)d_in[2];
  const int*   toks    = (const int*)d_in[3];
  const int*   ids     = (const int*)d_in[4];
  const float* tokemb  = (const float*)d_in[5];
  const float* Wq  = (const float*)d_in[6];
  const float* Wk  = (const float*)d_in[7];
  const float* Wv  = (const float*)d_in[8];
  const float* Wo  = (const float*)d_in[9];
  const float* W1  = (const float*)d_in[10];
  const float* b1  = (const float*)d_in[11];
  const float* W2  = (const float*)d_in[12];
  const float* b2  = (const float*)d_in[13];
  const float* Wlm = (const float*)d_in[14];
  const float* blm = (const float*)d_in[15];
  const float* Uw1 = (const float*)d_in[16];
  const float* Ub1 = (const float*)d_in[17];
  const float* Uw2 = (const float*)d_in[18];
  const float* Ub2 = (const float*)d_in[19];
  const float* Fw1 = (const float*)d_in[20];
  const float* Fb1 = (const float*)d_in[21];
  const float* Fw2 = (const float*)d_in[22];
  const float* Fb2 = (const float*)d_in[23];
  const float* iemb = (const float*)d_in[24];
  float* out = (float*)d_out;

  char* wsb = (char*)d_ws;
  size_t off = 0;
  auto alloc = [&](size_t bytes) -> void* {
    void* p = wsb + off;
    off += bytes;
    off = (off + 255) & ~(size_t)255;
    return p;
  };

  // transposed bf16 weights (N x K)
  __bf16* wqT  = (__bf16*)alloc((size_t)D * D * 2);
  __bf16* wkT  = (__bf16*)alloc((size_t)D * D * 2);
  __bf16* wvT  = (__bf16*)alloc((size_t)D * D * 2);
  __bf16* woT  = (__bf16*)alloc((size_t)D * D * 2);
  __bf16* w1T  = (__bf16*)alloc((size_t)D * FF * 2);
  __bf16* w2T  = (__bf16*)alloc((size_t)FF * D * 2);
  __bf16* wlmT = (__bf16*)alloc((size_t)D * V * 2);
  __bf16* uw1T = (__bf16*)alloc((size_t)U * D * 2);
  __bf16* uw2T = (__bf16*)alloc((size_t)D * D * 2);
  __bf16* fw1T = (__bf16*)alloc((size_t)2 * D * D * 2);
  __bf16* fw2T = (__bf16*)alloc((size_t)D * D * 2);
  __bf16* egt  = (__bf16*)alloc((size_t)Bn * D * 2);

  float*  hist = (float*)alloc((size_t)Bn * S * D * 4);
  float*  comb = (float*)alloc((size_t)Bn * 65 * D * 4);
  float*  l2   = (float*)alloc((size_t)Bn * 65 * D * 4);
  float*  seq  = (float*)alloc((size_t)Bn * 72 * D * 4);
  float*  l3   = (float*)alloc((size_t)Bn * 72 * D * 4);
  __bf16* qb   = (__bf16*)alloc((size_t)Bn * 72 * D * 2);
  __bf16* kb   = (__bf16*)alloc((size_t)Bn * 72 * D * 2);
  __bf16* vb   = (__bf16*)alloc((size_t)Bn * 72 * D * 2);
  float*  attn = (float*)alloc((size_t)Bn * 72 * D * 4);
  float*  hb   = (float*)alloc((size_t)Bn * 72 * D * 4);
  float*  ffb  = (float*)alloc((size_t)Bn * 72 * FF * 4);
  float*  aa   = (float*)alloc((size_t)Bn * D * 4);
  float*  tth  = (float*)alloc((size_t)Bn * D * 4);
  float*  tte  = (float*)alloc((size_t)Bn * D * 4);
  float*  fin  = (float*)alloc((size_t)Bn * 2 * D * 4);
  float*  f1b  = (float*)alloc((size_t)Bn * D * 4);
  float*  fusd = (float*)alloc((size_t)Bn * D * 4);
  float*  ctx  = (float*)alloc((size_t)Bn * T * D * 4);
  float*  lml  = (float*)alloc((size_t)Bn * T * V * 4);
  float*  ttl  = (float*)alloc((size_t)Bn * Bn * 4);

  auto cvtT = [&](const float* s, __bf16* d, int K, int N) {
    cvtT_bf16_kernel<<<(K * N + 255) / 256, 256, 0, stream>>>(s, d, K, N);
  };
  cvtT(Wq, wqT, D, D);
  cvtT(Wk, wkT, D, D);
  cvtT(Wv, wvT, D, D);
  cvtT(Wo, woT, D, D);
  cvtT(W1, w1T, D, FF);
  cvtT(W2, w2T, FF, D);
  cvtT(Wlm, wlmT, D, V);
  cvtT(Uw1, uw1T, U, D);
  cvtT(Uw2, uw2T, D, D);
  cvtT(Fw1, fw1T, 2 * D, D);
  cvtT(Fw2, fw2T, D, D);

  // mode: 0 plain f32, 1 relu+bias f32, 2 res(+bias) f32, 3 bf16-out
  auto gemm = [&](const float* Am, const __bf16* WTm, const float* bias,
                  const float* resp, void* Cm, int M, int N, int K, int mode) {
    dim3 g(N / 128, M / 128);
    switch (mode) {
      case 1:
        gemm_kernel<true, false, false>
            <<<g, 256, 0, stream>>>(Am, WTm, bias, nullptr, Cm, M, N, K);
        break;
      case 2:
        gemm_kernel<false, true, false>
            <<<g, 256, 0, stream>>>(Am, WTm, bias, resp, Cm, M, N, K);
        break;
      case 3:
        gemm_kernel<false, false, true>
            <<<g, 256, 0, stream>>>(Am, WTm, nullptr, nullptr, Cm, M, N, K);
        break;
      default:
        gemm_kernel<false, false, false>
            <<<g, 256, 0, stream>>>(Am, WTm, bias, nullptr, Cm, M, N, K);
    }
  };

  const float scale = 0.044194173824159216f;  // 1/sqrt(512)

  auto run_layer = [&](const float* x, int L, float* outb) {
    const int M = Bn * L;
    gemm(x, wqT, nullptr, nullptr, qb, M, D, D, 3);
    gemm(x, wkT, nullptr, nullptr, kb, M, D, D, 3);
    gemm(x, wvT, nullptr, nullptr, vb, M, D, D, 3);
    attention_kernel<<<Bn, 256, 0, stream>>>(qb, kb, vb, attn, L, scale);
    gemm(attn, woT, nullptr, x, hb, M, D, D, 2);    // h = x + attn@Wo
    gemm(hb, w1T, b1, nullptr, ffb, M, FF, D, 1);   // relu(h@W1 + b1)
    gemm(ffb, w2T, b2, hb, outb, M, D, FF, 2);      // h + ff@W2 + b2
  };

  // layer 1: history
  run_layer(hist_in, S, hist);

  // layer 2: history + action token, take last row
  int n = Bn * 65 * D;
  build_combined_kernel<<<(n + 255) / 256, 256, 0, stream>>>(hist, actenc,
                                                             comb, n);
  run_layer(comb, 65, l2);
  n = Bn * D;
  extract_rows_kernel<<<(n + 255) / 256, 256, 0, stream>>>(l2, aa, 65, 64, 1,
                                                           n);

  // user tower + fusion MLP
  gemm(userf, uw1T, Ub1, nullptr, tth, Bn, D, U, 1);
  gemm(tth, uw2T, Ub2, nullptr, tte, Bn, D, D, 0);
  n = Bn * 2 * D;
  concat2_kernel<<<(n + 255) / 256, 256, 0, stream>>>(aa, tte, fin, D, D, n);
  gemm(fin, fw1T, Fb1, nullptr, f1b, Bn, D, 2 * D, 1);
  gemm(f1b, fw2T, Fb2, nullptr, fusd, Bn, D, D, 0);

  // layer 3: history + token embeddings
  n = Bn * 72 * D;
  build_seq_kernel<<<(n + 255) / 256, 256, 0, stream>>>(hist, tokemb, toks,
                                                        seq, n);
  run_layer(seq, 72, l3);
  n = Bn * T * D;
  extract_rows_kernel<<<(n + 255) / 256, 256, 0, stream>>>(l3, ctx, 72, S - 1,
                                                           T, n);

  // LM head + generation loss
  gemm(ctx, wlmT, blm, nullptr, lml, Bn * T, V, D, 0);
  zero_out_kernel<<<1, 1, 0, stream>>>(out);
  nll_rows_kernel<<<Bn * T, 256, 0, stream>>>(lml, toks, V, 1.0f / Bn, out);

  // retrieval loss
  n = Bn * D;
  gather_items_kernel<<<(n + 255) / 256, 256, 0, stream>>>(iemb, ids, egt);
  gemm(fusd, egt, nullptr, nullptr, ttl, Bn, Bn, D, 0);
  nll_rows_kernel<<<Bn, 256, 0, stream>>>(ttl, nullptr, Bn, 1.0f / Bn, out);
}
